// ScenarioExtractionLayer_20426864460426
// MI455X (gfx1250) — matrix operand order repaired and checked by reading, not compiled, not run on hardware
//
#include <hip/hip_runtime.h>
#include <hip/hip_bf16.h>

typedef _Float16 h8   __attribute__((ext_vector_type(8)));
typedef _Float16 v16h __attribute__((ext_vector_type(16)));
typedef float    v8f  __attribute__((ext_vector_type(8)));

#define TILE_M 128
#define XA_ST  296   // 288 cols + pad, halfs (rows 16B-aligned)
#define H_ST   136   // 128 cols + pad, halfs

// ---- workspace offsets (in halfs). total = 1,120,256 halfs = ~2.24 MB ----
#define OFF_SW1  0u        // 8 experts * KS8 * NB8 * 512 = 262144
#define OFF_SW2  262144u   // 8 * 4*8*512 = 131072
#define OFF_SW3  393216u
#define OFF_PW1  524288u   // 8 * 9*8*512 = 294912
#define OFF_PW2  819200u
#define OFF_PW3  950272u
#define OFF_AWG1 1081344u  // 9*8*512 = 36864  (aw1 || gw1 combined, [288][128])
#define OFF_W2C  1118208u  // 4*1*512 = 2048   (aw2/gw2 combined, [128][16])

// ============ prep: f32 -> f16, swizzled to WMMA B-fragment order ============
// B (32x16) k order used here, A matches: lane l holds col N=l%16, K=(l/16)*16+h, h=0..15 contiguous.
__global__ void swz_w(const float* __restrict__ src, _Float16* __restrict__ dst,
                      int K, int N, int KS, int NB) {
    int bid = blockIdx.x;                 // e*NB*KS + nb*KS + ks
    int blocksPerE = NB * KS;
    int e  = bid / blocksPerE;
    int r  = bid - e * blocksPerE;
    int nb = r / KS;
    int ks = r - nb * KS;
    int t = threadIdx.x;                  // 512 threads
    int lane = t >> 4, h = t & 15;
    int row = ks * 32 + (lane >> 4) * 16 + h;
    int col = nb * 16 + (lane & 15);
    dst[(size_t)bid * 512 + t] = (_Float16)src[(size_t)e * K * N + (size_t)row * N + col];
}

// combined attention/gate layer-1 weight: [288][128], cols 0..63 = aw1, 64..127 = gw1
__global__ void swz_awg1(const float* __restrict__ aw1, const float* __restrict__ gw1,
                         _Float16* __restrict__ dst) {
    const int KS = 9;
    int bid = blockIdx.x;                 // nb*KS + ks  (72 blocks)
    int nb = bid / KS, ks = bid % KS;
    int t = threadIdx.x;
    int lane = t >> 4, h = t & 15;
    int row = ks * 32 + (lane >> 4) * 16 + h;   // 0..287
    int col = nb * 16 + (lane & 15);            // 0..127
    float v = (col < 64) ? aw1[row * 64 + col] : gw1[row * 64 + (col - 64)];
    dst[(size_t)bid * 512 + t] = (_Float16)v;
}

// combined layer-2: [128][16]; rows<64 & cols<8 -> aw2, rows>=64 & cols 8..9 -> gw2, else 0
__global__ void swz_w2c(const float* __restrict__ aw2, const float* __restrict__ gw2,
                        _Float16* __restrict__ dst) {
    int ks = blockIdx.x;                  // 4 blocks, nb=0
    int t = threadIdx.x;
    int lane = t >> 4, h = t & 15;
    int row = ks * 32 + (lane >> 4) * 16 + h;   // 0..127
    int col = lane & 15;                        // 0..15
    float v = 0.0f;
    if (row < 64 && col < 8)                    v = aw2[row * 8 + col];
    else if (row >= 64 && col >= 8 && col < 10) v = gw2[(row - 64) * 2 + (col - 8)];
    dst[(size_t)ks * 512 + t] = (_Float16)v;
}

// ============================ main fused kernel =============================
__device__ __forceinline__ v16h ldA(const _Float16* p) {
    // A 16x32 f16, k order used here (as B): lane l row M=l%16; halfs 0..15 -> K = g*16+0..15
    union { v16h v; h8 q[2]; } u;
    u.q[0] = *(const h8*)p;
    u.q[1] = *(const h8*)(p + 8);
    return u.v;
}
__device__ __forceinline__ v16h ldB(const _Float16* p) {
    union { v16h v; h8 q[2]; } u;
    u.q[0] = *(const h8*)p;
    u.q[1] = *(const h8*)(p + 8);
    return u.v;
}

// Wave owns one 16-col block; computes all 8 row-blocks (128 rows) of that column strip.
// B fragment loaded once per k-step (L2-resident, coalesced b128).
__device__ __forceinline__ void gemm_cols(const _Float16* __restrict__ A, int aStride,
                                          const _Float16* __restrict__ Wnb, int KS,
                                          int lrow, int g, int lane, v8f c[8]) {
    #pragma unroll
    for (int rb = 0; rb < 8; ++rb)
        #pragma unroll
        for (int v = 0; v < 8; ++v) c[rb][v] = 0.0f;
    const _Float16* arow = A + (size_t)lrow * aStride + g * 16;
    const _Float16* bp   = Wnb + lane * 16;
    for (int ks = 0; ks < KS; ++ks) {
        v16h b = ldB(bp + ks * 512);
        const _Float16* ap = arow + ks * 32;
        #pragma unroll
        for (int rb = 0; rb < 8; ++rb) {
            v16h a = ldA(ap + (size_t)(rb * 16) * aStride);
            c[rb] = __builtin_amdgcn_wmma_f32_16x16x32_f16(
                false, a, false, b, (short)0, c[rb], false, false);
        }
    }
}

__device__ __forceinline__ void store_hidden(v8f c[8], float bv, _Float16* __restrict__ D,
                                             int col, int g) {
    #pragma unroll
    for (int rb = 0; rb < 8; ++rb) {
        _Float16* dp = D + (size_t)(rb * 16 + g * 8) * H_ST + col;
        #pragma unroll
        for (int v = 0; v < 8; ++v) {
            float h = c[rb][v] + bv;
            h = fmaxf(h, 0.0f);
            dp[(size_t)v * H_ST] = (_Float16)h;
        }
    }
}

__device__ __forceinline__ void acc_final(v8f c[8], float bv, const float* __restrict__ scl,
                                          int e, int g, float accOut[8][8]) {
    #pragma unroll
    for (int rb = 0; rb < 8; ++rb) {
        int row0 = rb * 16 + g * 8;
        #pragma unroll
        for (int v = 0; v < 8; ++v) {
            float s = scl[(row0 + v) * 8 + e];   // LDS broadcast across lanes 0-15 / 16-31
            accOut[rb][v] += s * (c[rb][v] + bv);
        }
    }
}

__global__ __launch_bounds__(256)
__attribute__((amdgpu_num_vgpr(256)))   // keep all VGPRs directly addressable: no s_set_vgpr_msb
void fused_moe(
    const float* __restrict__ x, const int* __restrict__ sids,
    const float* __restrict__ emb,
    const float* __restrict__ sb1, const float* __restrict__ sb2, const float* __restrict__ sb3,
    const float* __restrict__ pb1, const float* __restrict__ pb2, const float* __restrict__ pb3,
    const float* __restrict__ ab1, const float* __restrict__ ab2,
    const float* __restrict__ gb1, const float* __restrict__ gb2,
    const _Float16* __restrict__ wz, float* __restrict__ out) {

    __shared__ __align__(16) _Float16 sXA[TILE_M * XA_ST];   // 75,776 B
    __shared__ __align__(16) _Float16 sHA[TILE_M * H_ST];    // 34,816 B
    __shared__ __align__(16) _Float16 sHB[TILE_M * H_ST];    // 34,816 B (also logits scratch)
    __shared__ __align__(16) float    sScl[2 * TILE_M * 8];  //  8,192 B
    float* sLog = (float*)sHB;

    const int tid  = threadIdx.x;
    const int wave = tid >> 5;
    const int lane = tid & 31;
    const int lrow = lane & 15;
    const int g    = lane >> 4;
    const int rowBase = blockIdx.x * TILE_M;
    const int colOwn  = wave * 16 + lrow;     // owned output column (0..127)

    // ---- stage xa = [x | emb[sid]] as f16 into LDS ----
    for (int i = tid; i < TILE_M * 64; i += 256) {
        int r = i >> 6, c4 = i & 63;
        float4 v = ((const float4*)(x + (size_t)(rowBase + r) * 256))[c4];
        _Float16* p = &sXA[(size_t)r * XA_ST + c4 * 4];
        p[0] = (_Float16)v.x; p[1] = (_Float16)v.y; p[2] = (_Float16)v.z; p[3] = (_Float16)v.w;
    }
    if (tid < TILE_M) {
        int s = sids[rowBase + tid];
        const float4* ep = (const float4*)(emb + (size_t)s * 32);
        _Float16* p = &sXA[(size_t)tid * XA_ST + 256];
        #pragma unroll
        for (int q = 0; q < 8; ++q) {
            float4 v = ep[q];
            p[q*4+0] = (_Float16)v.x; p[q*4+1] = (_Float16)v.y;
            p[q*4+2] = (_Float16)v.z; p[q*4+3] = (_Float16)v.w;
        }
    }
    __syncthreads();

    v8f c[8];

    // ---- attention + gate hidden: relu(xa @ [aw1|gw1] + [ab1|gb1]) -> sHA ----
    gemm_cols(sXA, XA_ST, wz + OFF_AWG1 + (size_t)wave * (9 * 512), 9, lrow, g, lane, c);
    {
        float bv = (colOwn < 64) ? ab1[colOwn] : gb1[colOwn - 64];
        store_hidden(c, bv, sHA, colOwn, g);
    }
    __syncthreads();

    // ---- logits: hidden @ W2c  (10 useful cols) -> sLog (overlays sHB) ----
    if (wave == 0) {
        gemm_cols(sHA, H_ST, wz + OFF_W2C, 4, lrow, g, lane, c);
        float bv = (lrow < 8) ? ab2[lrow] : ((lrow < 10) ? gb2[lrow - 8] : 0.0f);
        #pragma unroll
        for (int rb = 0; rb < 8; ++rb)
            #pragma unroll
            for (int v = 0; v < 8; ++v)
                sLog[(rb * 16 + g * 8 + v) * 16 + lrow] = c[rb][v] + bv;
    }
    __syncthreads();

    // ---- per-row softmax(attn 8) and softmax(gate 2); fold into scales ----
    if (tid < TILE_M) {
        float lg[10];
        #pragma unroll
        for (int i = 0; i < 10; ++i) lg[i] = sLog[tid * 16 + i];
        float m = lg[0];
        #pragma unroll
        for (int i = 1; i < 8; ++i) m = fmaxf(m, lg[i]);
        float ex[8], den = 0.0f;
        #pragma unroll
        for (int i = 0; i < 8; ++i) { ex[i] = __expf(lg[i] - m); den += ex[i]; }
        float inv = 1.0f / den;
        float gm = fmaxf(lg[8], lg[9]);
        float e0 = __expf(lg[8] - gm), e1 = __expf(lg[9] - gm);
        float gi = 1.0f / (e0 + e1);
        float g0 = e0 * gi, g1 = e1 * gi;
        #pragma unroll
        for (int i = 0; i < 8; ++i) {
            float a = ex[i] * inv;
            sScl[tid * 8 + i]              = a * g0;   // weight for shared experts
            sScl[TILE_M * 8 + tid * 8 + i] = a * g1;   // weight for specific experts
        }
    }
    __syncthreads();

    // ---- expert loop: out += scl_sh*shared_e + scl_sp*specific_e ----
    float accOut[8][8];
    #pragma unroll
    for (int rb = 0; rb < 8; ++rb)
        #pragma unroll
        for (int v = 0; v < 8; ++v) accOut[rb][v] = 0.0f;

    #pragma unroll 1
    for (int e = 0; e < 8; ++e) {
        // shared expert
        gemm_cols(sXA, XA_ST, wz + OFF_SW1 + (size_t)e * 32768 + (size_t)wave * (8 * 512), 8, lrow, g, lane, c);
        store_hidden(c, sb1[e * 128 + colOwn], sHA, colOwn, g);
        __syncthreads();
        gemm_cols(sHA, H_ST, wz + OFF_SW2 + (size_t)e * 16384 + (size_t)wave * (4 * 512), 4, lrow, g, lane, c);
        store_hidden(c, sb2[e * 128 + colOwn], sHB, colOwn, g);
        __syncthreads();
        gemm_cols(sHB, H_ST, wz + OFF_SW3 + (size_t)e * 16384 + (size_t)wave * (4 * 512), 4, lrow, g, lane, c);
        acc_final(c, sb3[e * 128 + colOwn], sScl, e, g, accOut);
        // specific expert (xa incl. scenario embedding, K=288)
        gemm_cols(sXA, XA_ST, wz + OFF_PW1 + (size_t)e * 36864 + (size_t)wave * (9 * 512), 9, lrow, g, lane, c);
        store_hidden(c, pb1[e * 128 + colOwn], sHA, colOwn, g);
        __syncthreads();
        gemm_cols(sHA, H_ST, wz + OFF_PW2 + (size_t)e * 16384 + (size_t)wave * (4 * 512), 4, lrow, g, lane, c);
        store_hidden(c, pb2[e * 128 + colOwn], sHB, colOwn, g);
        __syncthreads();
        gemm_cols(sHB, H_ST, wz + OFF_PW3 + (size_t)e * 16384 + (size_t)wave * (4 * 512), 4, lrow, g, lane, c);
        acc_final(c, pb3[e * 128 + colOwn], sScl + TILE_M * 8, e, g, accOut);
    }

    // ---- write output [B,128] f32 ----
    const size_t outBase = (size_t)rowBase * 128;
    #pragma unroll
    for (int rb = 0; rb < 8; ++rb)
        #pragma unroll
        for (int v = 0; v < 8; ++v)
            out[outBase + (size_t)(rb * 16 + g * 8 + v) * 128 + colOwn] = accOut[rb][v];
}

// ================================ launch ====================================
extern "C" void kernel_launch(void* const* d_in, const int* in_sizes, int n_in,
                              void* d_out, int out_size, void* d_ws, size_t ws_size,
                              hipStream_t stream) {
    (void)in_sizes; (void)n_in; (void)out_size; (void)ws_size;
    const float* x   = (const float*)d_in[0];
    const int*   sid = (const int*)  d_in[1];
    const float* emb = (const float*)d_in[2];
    const float* sw1 = (const float*)d_in[3];
    const float* sb1 = (const float*)d_in[4];
    const float* sw2 = (const float*)d_in[5];
    const float* sb2 = (const float*)d_in[6];
    const float* sw3 = (const float*)d_in[7];
    const float* sb3 = (const float*)d_in[8];
    const float* pw1 = (const float*)d_in[9];
    const float* pb1 = (const float*)d_in[10];
    const float* pw2 = (const float*)d_in[11];
    const float* pb2 = (const float*)d_in[12];
    const float* pw3 = (const float*)d_in[13];
    const float* pb3 = (const float*)d_in[14];
    const float* aw1 = (const float*)d_in[15];
    const float* ab1 = (const float*)d_in[16];
    const float* aw2 = (const float*)d_in[17];
    const float* ab2 = (const float*)d_in[18];
    const float* gw1 = (const float*)d_in[19];
    const float* gb1 = (const float*)d_in[20];
    const float* gw2 = (const float*)d_in[21];
    const float* gb2 = (const float*)d_in[22];
    _Float16* wz = (_Float16*)d_ws;   // needs ~2.25 MB

    // weight prep (deterministic, every call; tiny vs main kernel)
    swz_w<<<8 * 8 * 8, 512, 0, stream>>>(sw1, wz + OFF_SW1, 256, 128, 8, 8);
    swz_w<<<8 * 8 * 4, 512, 0, stream>>>(sw2, wz + OFF_SW2, 128, 128, 4, 8);
    swz_w<<<8 * 8 * 4, 512, 0, stream>>>(sw3, wz + OFF_SW3, 128, 128, 4, 8);
    swz_w<<<8 * 8 * 9, 512, 0, stream>>>(pw1, wz + OFF_PW1, 288, 128, 9, 8);
    swz_w<<<8 * 8 * 4, 512, 0, stream>>>(pw2, wz + OFF_PW2, 128, 128, 4, 8);
    swz_w<<<8 * 8 * 4, 512, 0, stream>>>(pw3, wz + OFF_PW3, 128, 128, 4, 8);
    swz_awg1<<<72, 512, 0, stream>>>(aw1, gw1, wz + OFF_AWG1);
    swz_w2c<<<4, 512, 0, stream>>>(aw2, gw2, wz + OFF_W2C);

    fused_moe<<<65536 / TILE_M, 256, 0, stream>>>(
        x, sid, emb, sb1, sb2, sb3, pb1, pb2, pb3,
        ab1, ab2, gb1, gb2, wz, (float*)d_out);
}
